// BaseNet_15942918602883
// MI455X (gfx1250) — compile-verified, matrix operand order checked by reading, not run on hardware
//
#include <hip/hip_runtime.h>
#include <hip/hip_bf16.h>
#include <math.h>

// ---------------------------------------------------------------------------
// Fused 2-node-GCN scorer for MI455X (gfx1250), bf16 WMMA path.
//   layer1: [2B,512] @ [512,128]  (dominant: 262k FLOP/sample)  -> WMMA bf16
//   node-mean, layer2: [B,128]@[128,128], layer3: [B,128]@[128,64] -> WMMA bf16
//   readout: 64-dot + sigmoid in f32.
// ---------------------------------------------------------------------------

typedef __attribute__((ext_vector_type(16))) __bf16 v16bf;
typedef __attribute__((ext_vector_type(8)))  float  v8f;
typedef __attribute__((ext_vector_type(4)))  float  v4f;

#define WS_W1T_OFF   131072   // after W_in^T  (512*128 bf16)
#define WS_W2T_OFF   163840   // after W1^T    (128*128 bf16)
#define WS_WEFF_OFF  180224   // after W2^T    ( 64*128 bf16)

// Load a 16-element bf16 WMMA fragment for one lane:
// elements 0..7  = K[k0+sub .. k0+sub+7], elements 8..15 = K[k0+16+sub .. +7]
// (matches the ISA 16-bit A/B 16x32 VGPR layout; sub = 8 for lanes 16..31)
__device__ __forceinline__ v16bf load_bf16_frag(const __bf16* rowBase, int k0, int sub) {
    v16bf v;
    __builtin_memcpy(&v, rowBase + k0 + sub, 16);
    __builtin_memcpy(reinterpret_cast<char*>(&v) + 16, rowBase + k0 + 16 + sub, 16);
    return v;
}

// Same fragment but sourced from f32 global data with on-the-fly bf16 convert.
// p must already include the +sub lane offset.
__device__ __forceinline__ v16bf load_a_from_f32(const float* p, int k0) {
    v4f f0 = *(const v4f*)(p + k0);
    v4f f1 = *(const v4f*)(p + k0 + 4);
    v4f f2 = *(const v4f*)(p + k0 + 16);
    v4f f3 = *(const v4f*)(p + k0 + 20);
    v16bf a;
#pragma unroll
    for (int i = 0; i < 4; ++i) {
        a[i]      = (__bf16)f0[i];
        a[4 + i]  = (__bf16)f1[i];
        a[8 + i]  = (__bf16)f2[i];
        a[12 + i] = (__bf16)f3[i];
    }
    return a;
}

// ---------------------------------------------------------------------------
// Prep: transpose weights to [n][k] bf16 (B-operand friendly) + fold W_out.
// ---------------------------------------------------------------------------
__global__ void prep_kernel(const float* __restrict__ W_in, const float* __restrict__ W1,
                            const float* __restrict__ W2, const float* __restrict__ W_out,
                            __bf16* __restrict__ winT, __bf16* __restrict__ w1T,
                            __bf16* __restrict__ w2T, float* __restrict__ weff) {
    int i = blockIdx.x * blockDim.x + threadIdx.x;
    if (i < 512 * 128) {                                   // W_in^T [128][512]
        int n = i >> 9, k = i & 511;
        winT[i] = (__bf16)W_in[k * 128 + n];
    } else if (i < 512 * 128 + 128 * 128) {                // W1^T [128][128]
        int e = i - 512 * 128; int n = e >> 7, k = e & 127;
        w1T[e] = (__bf16)W1[k * 128 + n];
    } else if (i < 512 * 128 + 128 * 128 + 64 * 128) {     // W2^T [64][128]
        int e = i - (512 * 128 + 128 * 128); int n = e >> 7, k = e & 127;
        w2T[e] = (__bf16)W2[k * 64 + n];
    } else if (i < 512 * 128 + 128 * 128 + 64 * 128 + 64) {
        int e = i - (512 * 128 + 128 * 128 + 64 * 128);
        weff[e] = W_out[e] + W_out[64 + e];                // concat halves fold
    }
}

// ---------------------------------------------------------------------------
// Main fused kernel: 256 threads = 8 wave32; one tile = 64 samples.
// ---------------------------------------------------------------------------
__global__ __launch_bounds__(256)
void gnn_fused_kernel(const float* __restrict__ X,        // [2B][512] == combined
                      const float* __restrict__ b_in, const float* __restrict__ b1,
                      const float* __restrict__ b2, const float* __restrict__ b_out,
                      const __bf16* __restrict__ winT, const __bf16* __restrict__ w1T,
                      const __bf16* __restrict__ w2T, const float* __restrict__ weff,
                      float* __restrict__ out, int numTiles) {
    __shared__ __align__(16) __bf16 sW[512 * 128];            // W_in^T, 128 KB
    __shared__ __align__(16) unsigned char sT[64 * 128 * 2];  // 16 KB reused scratch

    const int tid  = threadIdx.x;
    const int w    = tid >> 5;          // wave id 0..7
    const int lane = tid & 31;
    const int l16  = lane & 15;
    const int hi   = lane >> 4;         // lane-half select
    const int sub  = hi << 3;           // K sub-offset for 16-bit fragments

    // Stage W_in^T into LDS once per block (reused across all tiles).
    {
        const uint4* src = (const uint4*)winT;
        uint4* dst = (uint4*)sW;
#pragma unroll
        for (int i = 0; i < 32; ++i) dst[tid + i * 256] = src[tid + i * 256];
    }
    __syncthreads();

    __bf16* sH  = (__bf16*)sT;   // [64][128] bf16 activations (layers 1..3 A-operand)
    float*  sG2 = (float*)sT;    // [64][64] f32 g2 (aliased, barrier-separated)

    for (int t = blockIdx.x; t < numTiles; t += gridDim.x) {
        // Speculative prefetch of the next tile this block will touch.
        {
            int tn = t + gridDim.x;
            if (tn < numTiles)
                __builtin_prefetch(X + (size_t)(tn * 128 + w * 16 + l16) * 512, 0, 1);
        }

        // ---- Layer 1: per wave, [16 x 512] @ [512 x 128] -> 128 WMMAs ----
        const float* xrow = X + (size_t)(t * 128 + w * 16 + l16) * 512 + sub;
        v8f acc1[8] = {};
        for (int k0 = 0; k0 < 512; k0 += 32) {
            v16bf a = load_a_from_f32(xrow, k0);
#pragma unroll
            for (int n = 0; n < 8; ++n) {
                v16bf b = load_bf16_frag(sW + (n * 16 + l16) * 512, k0, sub);
                acc1[n] = __builtin_amdgcn_wmma_f32_16x16x32_bf16(
                    false, a, false, b, (short)0, acc1[n], false, false);
            }
        }
        // bias + relu + mean over the two graph nodes (adjacent C rows),
        // write bf16 rows (8 samples per wave) into LDS.
#pragma unroll
        for (int n = 0; n < 8; ++n) {
            float bi = b_in[n * 16 + l16];
#pragma unroll
            for (int j = 0; j < 4; ++j) {
                float h0 = fmaxf(acc1[n][2 * j]     + bi, 0.f);
                float h1 = fmaxf(acc1[n][2 * j + 1] + bi, 0.f);
                sH[(w * 8 + hi * 4 + j) * 128 + n * 16 + l16] = (__bf16)(0.5f * (h0 + h1));
            }
        }
        __syncthreads();

        // ---- Layer 2: [64 x 128] @ [128 x 128], 32 (M,N) pairs over 8 waves ----
        {
            const int m  = w >> 1;
            const int nb = (w & 1) * 4;
            v8f acc2[4] = {};
            const __bf16* aRow = sH + (m * 16 + l16) * 128;
            for (int k0 = 0; k0 < 128; k0 += 32) {
                v16bf a = load_bf16_frag(aRow, k0, sub);
#pragma unroll
                for (int i = 0; i < 4; ++i) {
                    v16bf b = load_bf16_frag(w1T + ((nb + i) * 16 + l16) * 128, k0, sub);
                    acc2[i] = __builtin_amdgcn_wmma_f32_16x16x32_bf16(
                        false, a, false, b, (short)0, acc2[i], false, false);
                }
            }
            __syncthreads();                       // all sH reads complete
#pragma unroll
            for (int i = 0; i < 4; ++i) {
                float bi = b1[(nb + i) * 16 + l16];
#pragma unroll
                for (int j = 0; j < 8; ++j) {
                    int row = m * 16 + j + hi * 8;
                    sH[row * 128 + (nb + i) * 16 + l16] = (__bf16)fmaxf(acc2[i][j] + bi, 0.f);
                }
            }
        }
        __syncthreads();

        // ---- Layer 3: [64 x 128] @ [128 x 64], 16 (M,N) pairs over 8 waves ----
        {
            const int m  = w >> 1;
            const int nb = (w & 1) * 2;
            v8f acc3[2] = {};
            const __bf16* aRow = sH + (m * 16 + l16) * 128;
            for (int k0 = 0; k0 < 128; k0 += 32) {
                v16bf a = load_bf16_frag(aRow, k0, sub);
#pragma unroll
                for (int i = 0; i < 2; ++i) {
                    v16bf b = load_bf16_frag(w2T + ((nb + i) * 16 + l16) * 128, k0, sub);
                    acc3[i] = __builtin_amdgcn_wmma_f32_16x16x32_bf16(
                        false, a, false, b, (short)0, acc3[i], false, false);
                }
            }
            __syncthreads();                       // all sH reads complete
#pragma unroll
            for (int i = 0; i < 2; ++i) {
                float bi = b2[(nb + i) * 16 + l16];
#pragma unroll
                for (int j = 0; j < 8; ++j) {
                    int row = m * 16 + j + hi * 8;
                    sG2[row * 64 + (nb + i) * 16 + l16] = fmaxf(acc3[i][j] + bi, 0.f);
                }
            }
        }
        __syncthreads();

        // ---- Readout: score = g2 . weff + b_out; sigmoid (no WMMA here) ----
        if (tid < 64) {
            float sc = b_out[0];
            const float* row = sG2 + tid * 64;
#pragma unroll 8
            for (int c = 0; c < 64; ++c) sc += row[c] * weff[c];
            out[t * 64 + tid] = 1.f / (1.f + __expf(-sc));
        }
        __syncthreads();   // protect sT reuse by next tile's layer-1 writes
    }
}

// ---------------------------------------------------------------------------
extern "C" void kernel_launch(void* const* d_in, const int* in_sizes, int n_in,
                              void* d_out, int out_size, void* d_ws, size_t ws_size,
                              hipStream_t stream) {
    const float* combined = (const float*)d_in[0];
    const float* W_in  = (const float*)d_in[1];
    const float* b_in  = (const float*)d_in[2];
    const float* W1    = (const float*)d_in[3];
    const float* b1    = (const float*)d_in[4];
    const float* W2    = (const float*)d_in[5];
    const float* b2    = (const float*)d_in[6];
    const float* W_out = (const float*)d_in[7];
    const float* b_out = (const float*)d_in[8];
    float* out = (float*)d_out;

    __bf16* winT = (__bf16*)d_ws;
    __bf16* w1T  = (__bf16*)((char*)d_ws + WS_W1T_OFF);
    __bf16* w2T  = (__bf16*)((char*)d_ws + WS_W2T_OFF);
    float*  weff = (float*)((char*)d_ws + WS_WEFF_OFF);

    const int B = in_sizes[0] / 1024;          // combined is [B, 1024]
    const int numTiles = B / 64;               // 64 samples per tile

    const int prepN = 512 * 128 + 128 * 128 + 64 * 128 + 64;
    prep_kernel<<<(prepN + 255) / 256, 256, 0, stream>>>(W_in, W1, W2, W_out,
                                                         winT, w1T, w2T, weff);

    int grid = numTiles < 512 ? numTiles : 512;
    if (grid < 1) grid = 1;
    gnn_fused_kernel<<<grid, 256, 0, stream>>>(combined, b_in, b1, b2, b_out,
                                               winT, w1T, w2T, weff, out, numTiles);
}
